// RGCNEncoder_78116865180359
// MI455X (gfx1250) — matrix operand order repaired and checked by reading, not compiled, not run on hardware
//
#include <hip/hip_runtime.h>
#include <stdint.h>

// ---------------------------------------------------------------------------
// RGCN encoder for MI455X (gfx1250, wave32, WMMA bf16 -> f32 accumulate)
//
// Restructuring: each RGCN conv  out = x@root + b + sum_r mean_r(x)@W[r]
// becomes ONE GEMM:  [N x 7*d_in] @ [7*d_in x d_out]  where the A matrix is
// [x | mean_0 | .. | mean_5] (built in bf16, mean scaling fused) and B is
// [root; W_0; ..; W_5] pre-transposed to bf16 B^T for contiguous LDS staging.
// GEMM uses v_wmma_f32_16x16x32_bf16 with the documented wave32 fragment
// layouts; bias + ReLU fused in the epilogue.
// ---------------------------------------------------------------------------

#define IN_C    128
#define OUT_C   256
#define HALF_C  128
#define NUM_REL 6

typedef __attribute__((ext_vector_type(16))) __bf16 v16bf;
typedef __attribute__((ext_vector_type(8)))  float  v8f;

__device__ __forceinline__ unsigned short f32_to_bf16(float f) {
  union { float f; uint32_t u; } c; c.f = f;
  uint32_t u = c.u;
  u += 0x7fffu + ((u >> 16) & 1u);   // round-to-nearest-even
  return (unsigned short)(u >> 16);
}

// ---------------------------------------------------------------------------
// Edge scatter: agg[dst][rel][c] += x[src][c] ; cnt[dst][rel] += 1
// One 128-thread block per edge; f32 global atomics.
// ---------------------------------------------------------------------------
__global__ __launch_bounds__(128)
void scatter_mean_accum(const float* __restrict__ X, const int* __restrict__ src,
                        const int* __restrict__ dst, const int* __restrict__ et,
                        float* __restrict__ agg, float* __restrict__ cnt,
                        int E, int d) {
  int e = blockIdx.x;
  if (e >= E) return;
  int s = src[e], dd = dst[e], r = et[e];
  const float* xs = X + (long long)s * d;
  float* ag = agg + ((long long)dd * NUM_REL + r) * d;
  for (int c = threadIdx.x; c < d; c += 128) atomicAdd(&ag[c], xs[c]);
  if (threadIdx.x == 0) atomicAdd(&cnt[(long long)dd * NUM_REL + r], 1.0f);
}

// ---------------------------------------------------------------------------
// Build bf16 A matrices (rows padded to Mpad with zeros).
// ---------------------------------------------------------------------------
__global__ __launch_bounds__(256)
void prep_a_dense(const float* __restrict__ X, unsigned short* __restrict__ A,
                  int Nn, int K, long long total) {
  long long i = (long long)blockIdx.x * 256 + threadIdx.x;
  if (i >= total) return;
  int row = (int)(i / K), col = (int)(i % K);
  float v = (row < Nn) ? X[(long long)row * K + col] : 0.0f;
  A[i] = f32_to_bf16(v);
}

__global__ __launch_bounds__(256)
void prep_a_rgcn(const float* __restrict__ X, const float* __restrict__ agg,
                 const float* __restrict__ cnt, unsigned short* __restrict__ A,
                 int Nn, int d, long long total) {
  long long i = (long long)blockIdx.x * 256 + threadIdx.x;
  if (i >= total) return;
  int K = d * (1 + NUM_REL);
  int row = (int)(i / K), col = (int)(i % K);
  float v = 0.0f;
  if (row < Nn) {
    if (col < d) {
      v = X[(long long)row * d + col];
    } else {
      int r = col / d - 1;
      int c = col - d * (r + 1);
      float cc = cnt[(long long)row * NUM_REL + r];
      v = agg[((long long)row * NUM_REL + r) * d + c] / fmaxf(cc, 1.0f);
    }
  }
  A[i] = f32_to_bf16(v);
}

// ---------------------------------------------------------------------------
// Build bf16 B^T [Ncols x K] from row-major fp32 weights.
// ---------------------------------------------------------------------------
__global__ __launch_bounds__(256)
void prep_bt_dense(const float* __restrict__ Wd, unsigned short* __restrict__ BT,
                   int K, int Ncols, long long total) {
  long long i = (long long)blockIdx.x * 256 + threadIdx.x;
  if (i >= total) return;
  int o = (int)(i / K), k = (int)(i % K);
  BT[i] = f32_to_bf16(Wd[(long long)k * Ncols + o]);
}

__global__ __launch_bounds__(256)
void prep_bt_rgcn(const float* __restrict__ root, const float* __restrict__ W,
                  unsigned short* __restrict__ BT, int d_in, int d_out,
                  long long total) {
  long long i = (long long)blockIdx.x * 256 + threadIdx.x;
  if (i >= total) return;
  int K = d_in * (1 + NUM_REL);
  int o = (int)(i / K), col = (int)(i % K);
  float v;
  if (col < d_in) {
    v = root[(long long)col * d_out + o];
  } else {
    int r = col / d_in - 1;
    int k = col - d_in * (r + 1);
    v = W[((long long)r * d_in + k) * (long long)d_out + o];
  }
  BT[i] = f32_to_bf16(v);
}

// ---------------------------------------------------------------------------
// bf16 GEMM with fused bias (+ optional ReLU).
//   out[M x Ncols] = A[M x K] @ B[K x Ncols]  (B given as B^T [Ncols x K])
// Block: 256 threads = 8 waves, tile 128(M) x 64(N). Each wave owns a
// 16 x 64 strip = 4 v_wmma_f32_16x16x32_bf16 accumulators.
// ---------------------------------------------------------------------------
#define BM 128
#define BN 64
#define BK 32
#define LDSA_STRIDE 40   // halves; 80B row pitch, 16B aligned

__global__ __launch_bounds__(256)
void gemm_bf16(const unsigned short* __restrict__ A,   // [Mpad x K] bf16
               const unsigned short* __restrict__ BT,  // [Ncols x K] bf16
               const float* __restrict__ bias,
               float* __restrict__ out,
               int Mn, int K, int ldout, int do_relu) {
  __shared__ unsigned short As[BM * LDSA_STRIDE];
  __shared__ unsigned short Bs[BN * LDSA_STRIDE];

  const int tid  = threadIdx.x;
  const int lane = tid & 31;
  const int wave = tid >> 5;      // 0..7  -> 16-row M strip within 128-row tile
  const int ln   = lane & 15;
  const int hsel = lane >> 4;     // lane half select
  const long long rowBase = (long long)blockIdx.x * BM;
  const long long colBase = (long long)blockIdx.y * BN;

  v8f acc[4];
  acc[0] = {}; acc[1] = {}; acc[2] = {}; acc[3] = {};

  for (int k0 = 0; k0 < K; k0 += BK) {
    // --- stage A tile: 128 rows x 32 halves = 512 uint4, 2 per thread
#pragma unroll
    for (int i = 0; i < 2; ++i) {
      int idx = tid + i * 256;
      int r = idx >> 2, c = idx & 3;
      *(uint4*)(&As[r * LDSA_STRIDE + c * 8]) =
          *(const uint4*)(A + (rowBase + r) * (long long)K + k0 + c * 8);
    }
    // --- stage B^T tile: 64 rows x 32 halves = 256 uint4, 1 per thread
    {
      int r = tid >> 2, c = tid & 3;
      *(uint4*)(&Bs[r * LDSA_STRIDE + c * 8]) =
          *(const uint4*)(BT + (colBase + r) * (long long)K + k0 + c * 8);
    }
    __syncthreads();

    // A fragment (ISA 16-bit A 16x32 layout: lanes 0-15 K{0..7,16..23},
    // lanes 16-31 K{8..15,24..31}); contiguous in row-major LDS -> 2x b128.
    const int ar = wave * 16 + ln;
    const int kb = hsel ? 8 : 0;
    union { v16bf v; uint4 u[2]; } af;
    af.u[0] = *(const uint4*)(&As[ar * LDSA_STRIDE + kb]);
    af.u[1] = *(const uint4*)(&As[ar * LDSA_STRIDE + kb + 16]);

    // B fragments (lanes 0-15: K{0..7,16..23}, lanes 16-31: K{8..15,24..31}, N=lane&15);
    // contiguous in B^T LDS rows -> 2x b128 each.
    const int kb2 = hsel ? 8 : 0;
#pragma unroll
    for (int nt = 0; nt < 4; ++nt) {
      union { v16bf v; uint4 u[2]; } bf;
      const int bc = (nt * 16 + ln) * LDSA_STRIDE + kb2;
      bf.u[0] = *(const uint4*)(&Bs[bc]);
      bf.u[1] = *(const uint4*)(&Bs[bc + 16]);
      acc[nt] = __builtin_amdgcn_wmma_f32_16x16x32_bf16(
          false, af.v, false, bf.v, (short)0, acc[nt], false, false);
    }
    __syncthreads();
  }

  // Epilogue per documented C/D layout: VGPR p -> M=p (lanes 0-15) / p+8.
#pragma unroll
  for (int nt = 0; nt < 4; ++nt) {
    const int col = (int)colBase + nt * 16 + ln;
    const float bv = bias[col];
#pragma unroll
    for (int p = 0; p < 8; ++p) {
      long long row = rowBase + wave * 16 + p + hsel * 8;
      if (row < Mn) {
        float v = acc[nt][p] + bv;
        if (do_relu) v = fmaxf(v, 0.0f);
        out[row * (long long)ldout + col] = v;
      }
    }
  }
}

// ---------------------------------------------------------------------------
// Orchestration
// ---------------------------------------------------------------------------
extern "C" void kernel_launch(void* const* d_in, const int* in_sizes, int n_in,
                              void* d_out, int out_size, void* d_ws, size_t ws_size,
                              hipStream_t stream) {
  const float* x       = (const float*)d_in[0];
  const int* edge_idx  = (const int*)  d_in[1];
  const int* edge_type = (const int*)  d_in[2];
  const float* enc_w0  = (const float*)d_in[3];
  const float* enc_b0  = (const float*)d_in[4];
  const float* enc_w1  = (const float*)d_in[5];
  const float* enc_b1  = (const float*)d_in[6];
  const float* enc_w2  = (const float*)d_in[7];
  const float* enc_b2  = (const float*)d_in[8];
  const float* W1      = (const float*)d_in[9];
  const float* root1   = (const float*)d_in[10];
  const float* b1      = (const float*)d_in[11];
  const float* W2      = (const float*)d_in[12];
  const float* root2   = (const float*)d_in[13];
  const float* b2      = (const float*)d_in[14];
  const float* W3      = (const float*)d_in[15];
  const float* root3   = (const float*)d_in[16];
  const float* b3      = (const float*)d_in[17];

  const int Nn = in_sizes[0] / IN_C;     // 50000
  const int E  = in_sizes[1] / 2;        // 800000
  const int* src = edge_idx;
  const int* dst = edge_idx + E;
  const int Mpad = ((Nn + BM - 1) / BM) * BM;
  const int Kmax = OUT_C * (1 + NUM_REL);   // 1792

  // ---- workspace carve (~595 MB total) ----
  char* w = (char*)d_ws;
  auto take = [&](size_t bytes) -> char* {
    char* p = w; w += (bytes + 255) & ~(size_t)255; return p;
  };
  unsigned short* Abuf  = (unsigned short*)take((size_t)Mpad * Kmax * 2);       // bf16 A
  unsigned short* BTbuf = (unsigned short*)take((size_t)OUT_C * Kmax * 2);      // bf16 B^T
  float* agg  = (float*)take((size_t)Nn * NUM_REL * OUT_C * 4);                 // scatter accum
  float* cnt  = (float*)take((size_t)Nn * NUM_REL * 4);                         // degree counts
  float* bufA = (float*)take((size_t)Nn * OUT_C * 4);                           // fp32 activations
  float* bufB = (float*)take((size_t)Nn * OUT_C * 4);
  float* outf = (float*)d_out;                                                  // [Nn x 256]

  const dim3 blk(256);
  auto cdiv = [](long long a, long long b) { return (unsigned)((a + b - 1) / b); };

  // ============ dense encoder branch ============
  // L0: h0 = relu(x @ enc_w0 + b0)   [N,128]->[N,256]
  { long long ta = (long long)Mpad * IN_C;
    prep_a_dense<<<cdiv(ta,256), blk, 0, stream>>>(x, Abuf, Nn, IN_C, ta);
    long long tb = (long long)IN_C * OUT_C;
    prep_bt_dense<<<cdiv(tb,256), blk, 0, stream>>>(enc_w0, BTbuf, IN_C, OUT_C, tb);
    gemm_bf16<<<dim3(Mpad/BM, OUT_C/BN), blk, 0, stream>>>(Abuf, BTbuf, enc_b0, bufA, Nn, IN_C, OUT_C, 1);
  }
  // L1: h1 = relu(h0 @ enc_w1 + b1)  [N,256]->[N,256]
  { long long ta = (long long)Mpad * OUT_C;
    prep_a_dense<<<cdiv(ta,256), blk, 0, stream>>>(bufA, Abuf, Nn, OUT_C, ta);
    long long tb = (long long)OUT_C * OUT_C;
    prep_bt_dense<<<cdiv(tb,256), blk, 0, stream>>>(enc_w1, BTbuf, OUT_C, OUT_C, tb);
    gemm_bf16<<<dim3(Mpad/BM, OUT_C/BN), blk, 0, stream>>>(Abuf, BTbuf, enc_b1, bufB, Nn, OUT_C, OUT_C, 1);
  }
  // L2: self_feat = h1 @ enc_w2 + b2 -> out[:, 0:128]
  { long long ta = (long long)Mpad * OUT_C;
    prep_a_dense<<<cdiv(ta,256), blk, 0, stream>>>(bufB, Abuf, Nn, OUT_C, ta);
    long long tb = (long long)OUT_C * HALF_C;
    prep_bt_dense<<<cdiv(tb,256), blk, 0, stream>>>(enc_w2, BTbuf, OUT_C, HALF_C, tb);
    gemm_bf16<<<dim3(Mpad/BM, HALF_C/BN), blk, 0, stream>>>(Abuf, BTbuf, enc_b2, outf, Nn, OUT_C, OUT_C, 0);
  }

  // ============ RGCN branch ============
  // conv1: g1 = relu(rgcn(x)) -> bufA   (d_in=128, d_out=256, K=896)
  { const int d = IN_C, dO = OUT_C, K = d * (1 + NUM_REL);
    hipMemsetAsync(agg, 0, (size_t)Nn * NUM_REL * d * 4, stream);
    hipMemsetAsync(cnt, 0, (size_t)Nn * NUM_REL * 4, stream);
    scatter_mean_accum<<<E, dim3(128), 0, stream>>>(x, src, dst, edge_type, agg, cnt, E, d);
    long long ta = (long long)Mpad * K;
    prep_a_rgcn<<<cdiv(ta,256), blk, 0, stream>>>(x, agg, cnt, Abuf, Nn, d, ta);
    long long tb = (long long)dO * K;
    prep_bt_rgcn<<<cdiv(tb,256), blk, 0, stream>>>(root1, W1, BTbuf, d, dO, tb);
    gemm_bf16<<<dim3(Mpad/BM, dO/BN), blk, 0, stream>>>(Abuf, BTbuf, b1, bufA, Nn, K, dO, 1);
  }
  // conv2: g2 = relu(rgcn(g1)) -> bufB  (d_in=256, d_out=256, K=1792)
  { const int d = OUT_C, dO = OUT_C, K = d * (1 + NUM_REL);
    hipMemsetAsync(agg, 0, (size_t)Nn * NUM_REL * d * 4, stream);
    hipMemsetAsync(cnt, 0, (size_t)Nn * NUM_REL * 4, stream);
    scatter_mean_accum<<<E, dim3(128), 0, stream>>>(bufA, src, dst, edge_type, agg, cnt, E, d);
    long long ta = (long long)Mpad * K;
    prep_a_rgcn<<<cdiv(ta,256), blk, 0, stream>>>(bufA, agg, cnt, Abuf, Nn, d, ta);
    long long tb = (long long)dO * K;
    prep_bt_rgcn<<<cdiv(tb,256), blk, 0, stream>>>(root2, W2, BTbuf, d, dO, tb);
    gemm_bf16<<<dim3(Mpad/BM, dO/BN), blk, 0, stream>>>(Abuf, BTbuf, b2, bufB, Nn, K, dO, 1);
  }
  // conv3: graph_feat = rgcn(g2) -> out[:, 128:256]  (d_out=128, no relu)
  { const int d = OUT_C, dO = HALF_C, K = d * (1 + NUM_REL);
    hipMemsetAsync(agg, 0, (size_t)Nn * NUM_REL * d * 4, stream);
    hipMemsetAsync(cnt, 0, (size_t)Nn * NUM_REL * 4, stream);
    scatter_mean_accum<<<E, dim3(128), 0, stream>>>(bufB, src, dst, edge_type, agg, cnt, E, d);
    long long ta = (long long)Mpad * K;
    prep_a_rgcn<<<cdiv(ta,256), blk, 0, stream>>>(bufB, agg, cnt, Abuf, Nn, d, ta);
    long long tb = (long long)dO * K;
    prep_bt_rgcn<<<cdiv(tb,256), blk, 0, stream>>>(root3, W3, BTbuf, d, dO, tb);
    gemm_bf16<<<dim3(Mpad/BM, dO/BN), blk, 0, stream>>>(Abuf, BTbuf, b3, outf + HALF_C, Nn, K, OUT_C, 0);
  }
}
